// fourier_UNet_conditional_300647711019
// MI455X (gfx1250) — matrix operand order repaired and checked by reading, not compiled, not run on hardware
//
#include <hip/hip_runtime.h>

typedef __attribute__((ext_vector_type(16))) _Float16 v16h;
typedef __attribute__((ext_vector_type(8)))  _Float16 v8h;
typedef __attribute__((ext_vector_type(4)))  _Float16 v4h;
typedef __attribute__((ext_vector_type(8)))  float    v8f;
typedef __attribute__((ext_vector_type(4)))  float    v4f;

// GCC-style int4 vector + address-space-qualified pointers for the gfx1250
// async global->LDS builtin (param types confirmed by compiler diagnostics).
typedef int v4i_g __attribute__((vector_size(16)));
typedef __attribute__((address_space(1))) v4i_g* gptr_v4i;
typedef __attribute__((address_space(3))) v4i_g* lptr_v4i;

#define TPB 256

#define GSL(i, n) for (long long i = blockIdx.x * (long long)blockDim.x + threadIdx.x; \
                       i < (n); i += (long long)gridDim.x * blockDim.x)

// gfx1250 async global->LDS copy (ASYNCcnt-tracked), guarded so the file
// compiles on toolchains without the builtin.
#if __has_builtin(__builtin_amdgcn_global_load_async_to_lds_b128)
#define ASYNC_A 1
#else
#define ASYNC_A 0
#endif

static __device__ inline void async_wait0()
{
#if __has_builtin(__builtin_amdgcn_s_wait_asynccnt)
    __builtin_amdgcn_s_wait_asynccnt(0);
#else
    asm volatile("s_wait_asynccnt 0x0" ::: "memory");
#endif
}

// ---------------------------------------------------------------------------
// WMMA GEMM:  C[z] = act( A[z] @ B[z or z%bmod] + bias )
// Row-major f32 in/out, f16 operands in WMMA, f32 accumulation via
// v_wmma_f32_16x16x32_f16.  128 threads = 4 waves in 2x2 tile arrangement;
// block tile = 32(M) x 64(N); each wave does two 16x16 WMMA tiles per k-step.
// A tile staged raw f32 via GLOBAL_LOAD_ASYNC_TO_LDS_B128 (if available),
// converted to f16 during fragment build; B staged transposed [n][k] as f16
// so each lane's B fragment is two separate 16B LDS loads (k kbB..kbB+7 and kbB+16..kbB+23) (2x ds_load_b128).
// ---------------------------------------------------------------------------
__global__ void __launch_bounds__(128)
gemm_wmma(const float* __restrict__ A, const float* __restrict__ Bm,
          const float* __restrict__ bias, float* __restrict__ C,
          int M, int N, int K, int lda, int ldb, int ldc,
          long long sA, long long sB, long long sC,
          int bmod, int relu)
{
#if ASYNC_A
    __shared__ float    As32[32 * 32];   // [m][k], f32
#else
    __shared__ _Float16 As[32 * 32];     // [m][k], f16
#endif
    __shared__ _Float16 Bs[64 * 32];     // transposed: [n][k], f16

    const int z = blockIdx.z;
    const long long aoff = (long long)z * sA;
    const long long boff = (long long)(bmod > 0 ? (z % bmod) : z) * sB;
    const long long coff = (long long)z * sC;

    const int m0  = blockIdx.x * 32;
    const int n00 = blockIdx.y * 64;
    const int t    = threadIdx.x;   // 0..127
    const int lane = t & 31;
    const int wave = t >> 5;        // 0..3
    const int wm = wave >> 1;       // 0..1 (M sub-tile)
    const int wn = wave & 1;        // 0..1 (N sub-tile)

    const bool mfull = (m0 + 32 <= M);
    const bool nfull = (n00 + 64 <= N);

    v8f acc0 = {}, acc1 = {};

    for (int k0 = 0; k0 < K; k0 += 32) {
        const bool kfull = (k0 + 32 <= K);

        // ---- stage A tile (32 x 32) ----
        if (mfull && kfull) {
#pragma unroll
            for (int j = 0; j < 2; ++j) {
                int idx4 = (t + j * 128) * 4;
                int r = idx4 >> 5, c = idx4 & 31;
                const float* src = A + aoff + (long long)(m0 + r) * lda + (k0 + c);
#if ASYNC_A
                float* srcnc = const_cast<float*>(src);
                __builtin_amdgcn_global_load_async_to_lds_b128(
                    (gptr_v4i)srcnc, (lptr_v4i)(&As32[r * 32 + c]), 0, 0);
#else
                v4f v = *(const v4f*)src;
                v4h h = { (_Float16)v.x, (_Float16)v.y, (_Float16)v.z, (_Float16)v.w };
                *(v4h*)(&As[r * 32 + c]) = h;
#endif
            }
        } else {
#pragma unroll
            for (int j = 0; j < 8; ++j) {
                int idx = t + j * 128;
                int r = idx >> 5, c = idx & 31;
                int rr = m0 + r, cc = k0 + c;
                int rcl = rr < M ? rr : M - 1;
                int ccl = cc < K ? cc : K - 1;
                float v = A[aoff + (long long)rcl * lda + ccl];
                float vz = (rr < M && cc < K) ? v : 0.f;
#if ASYNC_A
                As32[r * 32 + c] = vz;
#else
                As[r * 32 + c] = (_Float16)vz;
#endif
            }
        }

        // ---- stage B tile (32 x 64), transposed into LDS ----
        if (nfull && kfull) {
#pragma unroll
            for (int j = 0; j < 4; ++j) {
                int idx4 = (t + j * 128) * 4;
                int r = idx4 >> 6, c = idx4 & 63;
                v4f v = *(const v4f*)(Bm + boff + (long long)(k0 + r) * ldb + (n00 + c));
                Bs[(c + 0) * 32 + r] = (_Float16)v.x;
                Bs[(c + 1) * 32 + r] = (_Float16)v.y;
                Bs[(c + 2) * 32 + r] = (_Float16)v.z;
                Bs[(c + 3) * 32 + r] = (_Float16)v.w;
            }
        } else {
#pragma unroll
            for (int j = 0; j < 16; ++j) {
                int idx = t + j * 128;
                int r = idx >> 6, c = idx & 63;
                int rr = k0 + r, cc = n00 + c;
                int rcl = rr < K ? rr : K - 1;
                int ccl = cc < N ? cc : N - 1;
                float v = Bm[boff + (long long)rcl * ldb + ccl];
                Bs[c * 32 + r] = (_Float16)((rr < K && cc < N) ? v : 0.f);
            }
        }

        // prefetch next K tile into cache while this one computes
        if (k0 + 32 < K) {
            __builtin_prefetch(A + aoff + (long long)(m0 + lane) * lda + (k0 + 32), 0, 1);
            __builtin_prefetch(Bm + boff + (long long)(k0 + 32 + lane) * ldb + n00, 0, 1);
        }
#if ASYNC_A
        async_wait0();
#endif
        __syncthreads();

        // ---- fragments (ISA wave32 layouts) ----
        const int mrow = lane & 15;
        const int kbA = (lane >= 16) ? 8 : 0;    // A: interleaved half-runs
        const int kbB = (lane >= 16) ? 8 : 0;    // B: interleaved half-runs (same as A)
#if ASYNC_A
        const float* ap = &As32[(wm * 16 + mrow) * 32 + kbA];
        v4f f0 = *(const v4f*)(ap);
        v4f f1 = *(const v4f*)(ap + 4);
        v4f f2 = *(const v4f*)(ap + 16);
        v4f f3 = *(const v4f*)(ap + 20);
        v16h af;
#pragma unroll
        for (int e = 0; e < 4; ++e) {
            af[e]      = (_Float16)f0[e];
            af[e + 4]  = (_Float16)f1[e];
            af[e + 8]  = (_Float16)f2[e];
            af[e + 12] = (_Float16)f3[e];
        }
#else
        v8h a0 = *(const v8h*)(&As[(wm * 16 + mrow) * 32 + kbA]);
        v8h a1 = *(const v8h*)(&As[(wm * 16 + mrow) * 32 + kbA + 16]);
        v16h af = __builtin_shufflevector(a0, a1, 0, 1, 2, 3, 4, 5, 6, 7,
                                          8, 9, 10, 11, 12, 13, 14, 15);
#endif
        v8h b00 = *(const v8h*)(&Bs[(wn * 32 + mrow) * 32 + kbB]);
        v8h b01 = *(const v8h*)(&Bs[(wn * 32 + mrow) * 32 + kbB + 16]);
        v16h bf0 = __builtin_shufflevector(b00, b01, 0, 1, 2, 3, 4, 5, 6, 7,
                                           8, 9, 10, 11, 12, 13, 14, 15);
        v8h b10 = *(const v8h*)(&Bs[(wn * 32 + 16 + mrow) * 32 + kbB]);
        v8h b11 = *(const v8h*)(&Bs[(wn * 32 + 16 + mrow) * 32 + kbB + 16]);
        v16h bf1 = __builtin_shufflevector(b10, b11, 0, 1, 2, 3, 4, 5, 6, 7,
                                           8, 9, 10, 11, 12, 13, 14, 15);

        acc0 = __builtin_amdgcn_wmma_f32_16x16x32_f16(false, af, false, bf0,
                                                      (short)0, acc0, false, false);
        acc1 = __builtin_amdgcn_wmma_f32_16x16x32_f16(false, af, false, bf1,
                                                      (short)0, acc1, false, false);
        __syncthreads();
    }

    // ---- store C ----
    const int n0 = n00 + wn * 32 + (lane & 15);
    const int n1 = n0 + 16;
    const int mb = m0 + wm * 16 + ((lane >= 16) ? 8 : 0);
    float bv0 = 0.f, bv1 = 0.f;
    if (bias) {
        if (n0 < N) bv0 = bias[n0];
        if (n1 < N) bv1 = bias[n1];
    }
#pragma unroll
    for (int r8 = 0; r8 < 8; ++r8) {
        int m = mb + r8;
        if (m < M) {
            if (n0 < N) {
                float val = acc0[r8] + bv0;
                if (relu) val = fmaxf(val, 0.f);
                C[coff + (long long)m * ldc + n0] = val;
            }
            if (n1 < N) {
                float val = acc1[r8] + bv1;
                if (relu) val = fmaxf(val, 0.f);
                C[coff + (long long)m * ldc + n1] = val;
            }
        }
    }
}

// ---------------------------------------------------------------------------
// DFT matrices.  F (2M x H): rows 0..M-1 = cos(2*pi*m*h/H), rows M..2M-1 =
// -sin(...) (rfft convention).  G (H x 2M): truncated-irfft synthesis matrix.
// ---------------------------------------------------------------------------
__global__ void dft_init(float* __restrict__ F, float* __restrict__ G, int H, int Mm)
{
    const long long n = (long long)2 * Mm * H;
    const float twopi = 6.283185307179586f;
    GSL(idx, n) {
        {
            int k = (int)(idx / H), h = (int)(idx % H);
            int m = (k < Mm) ? k : k - Mm;
            float ang = twopi * (float)m * (float)h / (float)H;
            F[idx] = (k < Mm) ? __cosf(ang) : -__sinf(ang);
        }
        {
            int h = (int)(idx / (2 * Mm)), k = (int)(idx % (2 * Mm));
            int m = (k < Mm) ? k : k - Mm;
            float ang = twopi * (float)m * (float)h / (float)H;
            float cm = (m == 0) ? ((k < Mm) ? 1.f : 0.f) : 2.f;
            G[idx] = cm * ((k < Mm) ? __cosf(ang) : -__sinf(ang)) / (float)H;
        }
    }
}

// Pack complex weights w[I,O,Mm,2] into per-mode real block [[Wr,Wi],[-Wi,Wr]]
// of shape (2I x 2O), contiguous per mode.
__global__ void pack_wblk(const float* __restrict__ w, float* __restrict__ Bb,
                          int I, int O, int Mm)
{
    const long long n = (long long)Mm * 2 * I * 2 * O;
    GSL(idx, n) {
        int co = (int)(idx % (2 * O));
        int r  = (int)((idx / (2 * O)) % (2 * I));
        int m  = (int)(idx / ((long long)(2 * O) * (2 * I)));
        int i = (r < I) ? r : r - I;
        int o = (co < O) ? co : co - O;
        int reim; float sgn = 1.f;
        if (r < I) { reim = (co < O) ? 0 : 1; }
        else       { reim = (co < O) ? 1 : 0; if (co < O) sgn = -1.f; }
        Bb[idx] = sgn * w[(((long long)i * O + o) * Mm + m) * 2 + reim];
    }
}

// Xf [B][2Mm][W*I] -> Ac [B][Mm][W][2I]  (row = [Re(i) | Im(i)])
__global__ void k_interleave_h(const float* __restrict__ Xf, float* __restrict__ Ac,
                               int B, int Mm, int W, int I)
{
    const long long n = (long long)B * Mm * W * 2 * I;
    GSL(i, n) {
        int ii = (int)(i % (2 * I)); long long r = i / (2 * I);
        int w = (int)(r % W); r /= W;
        int m = (int)(r % Mm); int b = (int)(r / Mm);
        int re = ii < I; int ci = re ? ii : ii - I;
        Ac[i] = Xf[((long long)b * 2 * Mm + (re ? m : m + Mm)) * ((long long)W * I)
                   + (long long)w * I + ci];
    }
}

// oh [B][Mm][W][2O] -> ohT [B][2Mm][W*O]  (Re rows then Im rows)
__global__ void k_reorder_h(const float* __restrict__ oh, float* __restrict__ ohT,
                            int B, int Mm, int W, int O)
{
    const long long n = (long long)B * 2 * Mm * W * O;
    GSL(i, n) {
        int o = (int)(i % O); long long r = i / O;
        int w = (int)(r % W); r /= W;
        int k = (int)(r % (2 * Mm)); int b = (int)(r / (2 * Mm));
        int m = (k < Mm) ? k : k - Mm;
        ohT[i] = oh[(((long long)b * Mm + m) * W + w) * (2 * O) + ((k < Mm) ? 0 : O) + o];
    }
}

// Xf2 [BH][2Mm][I] -> Ac2 [Mm][BH][2I]
__global__ void k_interleave_w(const float* __restrict__ Xf2, float* __restrict__ Ac2,
                               long long BH, int Mm, int I)
{
    const long long n = BH * Mm * 2 * I;
    GSL(i, n) {
        int ii = (int)(i % (2 * I)); long long r = i / (2 * I);
        long long bh = r % BH; int m = (int)(r / BH);
        int re = ii < I; int ci = re ? ii : ii - I;
        Ac2[i] = Xf2[bh * (2LL * Mm * I) + (long long)(re ? m : m + Mm) * I + ci];
    }
}

// oh2 [Mm][BH][2O] -> oh2T [BH][2Mm][O]
__global__ void k_reorder_w(const float* __restrict__ oh2, float* __restrict__ oh2T,
                            long long BH, int Mm, int O)
{
    const long long n = BH * 2 * Mm * O;
    GSL(i, n) {
        int o = (int)(i % O); long long r = i / O;
        int k = (int)(r % (2 * Mm)); long long bh = r / (2 * Mm);
        int m = (k < Mm) ? k : k - Mm;
        oh2T[i] = oh2[((long long)m * BH + bh) * (2 * O) + ((k < Mm) ? 0 : O) + o];
    }
}

// out = relu(a + b + c); optional FiLM: out = gam[b]*out + bet[b]
__global__ void k_add3_relu_film(const float* __restrict__ a, const float* __restrict__ b,
                                 const float* __restrict__ c, float* __restrict__ out,
                                 long long n, long long perB,
                                 const float* __restrict__ gam,
                                 const float* __restrict__ bet, int gcol)
{
    GSL(i, n) {
        float s = a[i] + b[i] + c[i];
        s = fmaxf(s, 0.f);
        if (gam) {
            int bi = (int)(i / perB);
            s = gam[bi * 4 + gcol] * s + bet[bi * 4 + gcol];
        }
        out[i] = s;
    }
}

__global__ void k_bn(float* __restrict__ x, const float* __restrict__ w,
                     const float* __restrict__ b, const float* __restrict__ rm,
                     const float* __restrict__ rv, int C, long long n)
{
    GSL(i, n) {
        int c = (int)(i % C);
        x[i] = (x[i] - rm[c]) * rsqrtf(rv[c] + 1e-5f) * w[c] + b[c];
    }
}

__global__ void k_maxpool2(const float* __restrict__ in, float* __restrict__ out,
                           int B, int H, int W, int C)
{
    const long long n = (long long)B * (H / 2) * (W / 2) * C;
    GSL(i, n) {
        int c = (int)(i % C); long long r = i / C;
        int ow = (int)(r % (W / 2)); r /= (W / 2);
        int oh = (int)(r % (H / 2)); int b = (int)(r / (H / 2));
        const float* p = in + (((long long)b * H + oh * 2) * W + ow * 2) * C + c;
        long long rs = (long long)W * C;
        float m01 = fmaxf(p[0], p[C]);
        float m23 = fmaxf(p[rs], p[rs + C]);
        out[i] = fmaxf(m01, m23);
    }
}

__global__ void k_upsample2(const float* __restrict__ in, float* __restrict__ out,
                            int B, int H, int W, int C)
{
    const int Ho = 2 * H, Wo = 2 * W;
    const long long n = (long long)B * Ho * Wo * C;
    GSL(i, n) {
        int c = (int)(i % C); long long r = i / C;
        int ox = (int)(r % Wo); r /= Wo;
        int oy = (int)(r % Ho); int b = (int)(r / Ho);
        float fy = oy * (float)(H - 1) / (float)(Ho - 1);
        float fx = ox * (float)(W - 1) / (float)(Wo - 1);
        int y0 = (int)fy, x0 = (int)fx;
        int y1 = (y0 + 1 < H) ? y0 + 1 : H - 1;
        int x1 = (x0 + 1 < W) ? x0 + 1 : W - 1;
        float wy = fy - (float)y0, wx = fx - (float)x0;
        const float* base = in + ((long long)b * H * W) * C + c;
        float v00 = base[((long long)y0 * W + x0) * C];
        float v01 = base[((long long)y0 * W + x1) * C];
        float v10 = base[((long long)y1 * W + x0) * C];
        float v11 = base[((long long)y1 * W + x1) * C];
        out[i] = (1.f - wy) * ((1.f - wx) * v00 + wx * v01)
               +        wy  * ((1.f - wx) * v10 + wx * v11);
    }
}

__global__ void k_concat(const float* __restrict__ a, const float* __restrict__ b,
                         float* __restrict__ out, long long pix, int C1, int C2)
{
    const int C = C1 + C2;
    const long long n = pix * C;
    GSL(i, n) {
        int c = (int)(i % C); long long p = i / C;
        out[i] = (c < C1) ? a[p * C1 + c] : b[p * C2 + (c - C1)];
    }
}

// ---------------------------------------------------------------------------
// Host-side orchestration
// ---------------------------------------------------------------------------
struct Ws { char* base; size_t off; };
static inline float* walloc(Ws& w, size_t elems)
{
    float* p = (float*)(w.base + w.off);
    w.off += ((elems * 4 + 255) & ~(size_t)255);
    return p;
}

static inline unsigned ewblocks(long long n)
{
    long long b = (n + TPB - 1) / TPB;
    if (b > 1048576) b = 1048576;
    if (b < 1) b = 1;
    return (unsigned)b;
}

static inline void launch_gemm(hipStream_t s, const float* A, const float* B,
                               const float* bias, float* C,
                               int M, int N, int K, int lda, int ldb, int ldc,
                               long long sA, long long sB, long long sC,
                               int batch, int bmod, int relu)
{
    dim3 g((unsigned)((M + 31) / 32), (unsigned)((N + 63) / 64), (unsigned)batch);
    gemm_wmma<<<g, dim3(128), 0, s>>>(A, B, bias, C, M, N, K, lda, ldb, ldc,
                                      sA, sB, sC, bmod, relu);
}

// One FNO spectral block: relu(irfftH(WH*rfftH(x)) + irfftW(WW*rfftW(x)) + x@Wres+b),
// then optional FiLM.  X:[B,H,H,I] -> OUT:[B,H,H,O].  All matmuls via WMMA.
static void spectral_block(hipStream_t s, Ws scratch,
                           const float* X, float* OUT,
                           const float* w1, const float* w2,
                           const float* wres, const float* bres,
                           int B, int H, int I, int O, int Mm,
                           const float* gam, const float* bet, int gcol)
{
    const int W = H;
    const int M2 = 2 * Mm;
    const long long BH = (long long)B * H;

    float* F    = walloc(scratch, (size_t)M2 * H);
    float* G    = walloc(scratch, (size_t)H * M2);
    float* Bb1  = walloc(scratch, (size_t)Mm * 2 * I * 2 * O);
    float* Bb2  = walloc(scratch, (size_t)Mm * 2 * I * 2 * O);
    float* Xf   = walloc(scratch, (size_t)B * M2 * W * I);
    float* Ac   = walloc(scratch, (size_t)B * Mm * W * 2 * I);
    float* oh   = walloc(scratch, (size_t)B * Mm * W * 2 * O);
    float* ohT  = walloc(scratch, (size_t)B * M2 * W * O);
    float* xh   = walloc(scratch, (size_t)B * H * W * O);
    float* Xf2  = walloc(scratch, (size_t)BH * M2 * I);
    float* Ac2  = walloc(scratch, (size_t)((long long)Mm * BH * 2 * I));
    float* oh2  = walloc(scratch, (size_t)((long long)Mm * BH * 2 * O));
    float* oh2T = walloc(scratch, (size_t)BH * M2 * O);
    float* xw   = walloc(scratch, (size_t)B * H * W * O);
    float* res  = walloc(scratch, (size_t)B * H * W * O);

    dft_init<<<ewblocks((long long)M2 * H), TPB, 0, s>>>(F, G, H, Mm);
    long long nb = (long long)Mm * 2 * I * 2 * O;
    pack_wblk<<<ewblocks(nb), TPB, 0, s>>>(w1, Bb1, I, O, Mm);
    pack_wblk<<<ewblocks(nb), TPB, 0, s>>>(w2, Bb2, I, O, Mm);

    // ---- transform along H ----
    launch_gemm(s, F, X, nullptr, Xf, M2, W * I, H, H, W * I, W * I,
                0, (long long)H * W * I, (long long)M2 * W * I, B, 0, 0);
    k_interleave_h<<<ewblocks((long long)B * Mm * W * 2 * I), TPB, 0, s>>>(Xf, Ac, B, Mm, W, I);
    launch_gemm(s, Ac, Bb1, nullptr, oh, W, 2 * O, 2 * I, 2 * I, 2 * O, 2 * O,
                (long long)W * 2 * I, (long long)4 * I * O, (long long)W * 2 * O,
                B * Mm, Mm, 0);
    k_reorder_h<<<ewblocks((long long)B * M2 * W * O), TPB, 0, s>>>(oh, ohT, B, Mm, W, O);
    launch_gemm(s, G, ohT, nullptr, xh, H, W * O, M2, M2, W * O, W * O,
                0, (long long)M2 * W * O, (long long)H * W * O, B, 0, 0);

    // ---- transform along W ----
    launch_gemm(s, F, X, nullptr, Xf2, M2, I, W, W, I, I,
                0, (long long)W * I, (long long)M2 * I, (int)BH, 0, 0);
    k_interleave_w<<<ewblocks(BH * Mm * 2 * I), TPB, 0, s>>>(Xf2, Ac2, BH, Mm, I);
    launch_gemm(s, Ac2, Bb2, nullptr, oh2, (int)BH, 2 * O, 2 * I, 2 * I, 2 * O, 2 * O,
                BH * 2 * I, (long long)4 * I * O, BH * 2 * O, Mm, 0, 0);
    k_reorder_w<<<ewblocks(BH * M2 * O), TPB, 0, s>>>(oh2, oh2T, BH, Mm, O);
    launch_gemm(s, G, oh2T, nullptr, xw, W, O, M2, M2, O, O,
                0, (long long)M2 * O, (long long)W * O, (int)BH, 0, 0);

    // ---- residual pointwise linear ----
    launch_gemm(s, X, wres, bres, res, (int)(BH * W), O, I, I, O, O,
                0, 0, 0, 1, 0, 0);

    const long long n = (long long)B * H * W * O;
    k_add3_relu_film<<<ewblocks(n), TPB, 0, s>>>(xh, xw, res, OUT, n,
                                                 (long long)H * W * O, gam, bet, gcol);
}

extern "C" void kernel_launch(void* const* d_in, const int* in_sizes, int n_in,
                              void* d_out, int out_size, void* d_ws, size_t ws_size,
                              hipStream_t stream)
{
    (void)in_sizes; (void)n_in; (void)out_size; (void)ws_size;
#define IN(i) ((const float*)d_in[(i)])
    const float* x    = IN(0);
    const float* cond = IN(1);

    struct Blk { const float *w1, *w2, *wr, *br; };
    auto blk = [&](int base) { return Blk{ IN(base), IN(base + 1), IN(base + 2), IN(base + 3) }; };
    Blk inc = blk(2),  dn1 = blk(6),  dn2 = blk(10), dn3 = blk(14), dn4 = blk(18);
    Blk up1 = blk(22), up2 = blk(26), up3 = blk(30), up4 = blk(34);

    const int B = 8;
    Ws pws{ (char*)d_ws, 0 };

    // persistent activations
    float* x1  = walloc(pws, (size_t)B * 256 * 256 * 64);
    float* x2  = walloc(pws, (size_t)B * 128 * 128 * 128);
    float* x3  = walloc(pws, (size_t)B * 64 * 64 * 256);
    float* x4  = walloc(pws, (size_t)B * 32 * 32 * 512);
    float* x5  = walloc(pws, (size_t)B * 16 * 16 * 512);
    float* p1  = walloc(pws, (size_t)B * 128 * 128 * 64);
    float* p2  = walloc(pws, (size_t)B * 64 * 64 * 128);
    float* p3  = walloc(pws, (size_t)B * 32 * 32 * 256);
    float* p4  = walloc(pws, (size_t)B * 16 * 16 * 512);
    float* upb = walloc(pws, (size_t)B * 256 * 256 * 64);   // biggest upsample output
    float* cat = walloc(pws, (size_t)B * 256 * 256 * 128);  // biggest concat output
    float* u   = walloc(pws, (size_t)B * 256 * 256 * 64);
    float* fh1 = walloc(pws, (size_t)B * 16);
    float* fh2 = walloc(pws, (size_t)B * 256);
    float* fh3 = walloc(pws, (size_t)B * 1024);
    float* gms = walloc(pws, (size_t)B * 4);
    float* bts = walloc(pws, (size_t)B * 4);

    Ws scratch = pws;   // by-value snapshot: reused (reset) per spectral block

    // ---- FiLM generator MLP (eval mode) ----
    launch_gemm(stream, cond, IN(40), IN(41), fh1, B, 16, 8, 8, 16, 16, 0, 0, 0, 1, 0, 1);
    launch_gemm(stream, fh1, IN(42), IN(43), fh2, B, 256, 16, 16, 256, 256, 0, 0, 0, 1, 0, 1);
    k_bn<<<ewblocks(B * 256), TPB, 0, stream>>>(fh2, IN(46), IN(47), IN(48), IN(49), 256, (long long)B * 256);
    launch_gemm(stream, fh2, IN(44), IN(45), fh3, B, 1024, 256, 256, 1024, 1024, 0, 0, 0, 1, 0, 1);
    k_bn<<<ewblocks(B * 1024), TPB, 0, stream>>>(fh3, IN(50), IN(51), IN(52), IN(53), 1024, (long long)B * 1024);
    launch_gemm(stream, fh3, IN(54), IN(55), gms, B, 4, 1024, 1024, 4, 4, 0, 0, 0, 1, 0, 0);
    launch_gemm(stream, fh3, IN(56), IN(57), bts, B, 4, 1024, 1024, 4, 4, 0, 0, 0, 1, 0, 0);

    // ---- encoder ----
    spectral_block(stream, scratch, x,  x1, inc.w1, inc.w2, inc.wr, inc.br, B, 256, 4,   64, 32, nullptr, nullptr, 0);
    k_maxpool2<<<ewblocks((long long)B * 128 * 128 * 64), TPB, 0, stream>>>(x1, p1, B, 256, 256, 64);
    spectral_block(stream, scratch, p1, x2, dn1.w1, dn1.w2, dn1.wr, dn1.br, B, 128, 64, 128, 32, gms, bts, 0);
    k_maxpool2<<<ewblocks((long long)B * 64 * 64 * 128), TPB, 0, stream>>>(x2, p2, B, 128, 128, 128);
    spectral_block(stream, scratch, p2, x3, dn2.w1, dn2.w2, dn2.wr, dn2.br, B, 64, 128, 256, 16, gms, bts, 1);
    k_maxpool2<<<ewblocks((long long)B * 32 * 32 * 256), TPB, 0, stream>>>(x3, p3, B, 64, 64, 256);
    spectral_block(stream, scratch, p3, x4, dn3.w1, dn3.w2, dn3.wr, dn3.br, B, 32, 256, 512, 8, gms, bts, 2);
    k_maxpool2<<<ewblocks((long long)B * 16 * 16 * 512), TPB, 0, stream>>>(x4, p4, B, 32, 32, 512);
    spectral_block(stream, scratch, p4, x5, dn4.w1, dn4.w2, dn4.wr, dn4.br, B, 16, 512, 512, 4, gms, bts, 3);

    // ---- decoder ----
    k_upsample2<<<ewblocks((long long)B * 32 * 32 * 512), TPB, 0, stream>>>(x5, upb, B, 16, 16, 512);
    k_concat<<<ewblocks((long long)B * 32 * 32 * 1024), TPB, 0, stream>>>(x4, upb, cat, (long long)B * 32 * 32, 512, 512);
    spectral_block(stream, scratch, cat, u, up1.w1, up1.w2, up1.wr, up1.br, B, 32, 1024, 256, 4, nullptr, nullptr, 0);

    k_upsample2<<<ewblocks((long long)B * 64 * 64 * 256), TPB, 0, stream>>>(u, upb, B, 32, 32, 256);
    k_concat<<<ewblocks((long long)B * 64 * 64 * 512), TPB, 0, stream>>>(x3, upb, cat, (long long)B * 64 * 64, 256, 256);
    spectral_block(stream, scratch, cat, u, up2.w1, up2.w2, up2.wr, up2.br, B, 64, 512, 128, 8, nullptr, nullptr, 0);

    k_upsample2<<<ewblocks((long long)B * 128 * 128 * 128), TPB, 0, stream>>>(u, upb, B, 64, 64, 128);
    k_concat<<<ewblocks((long long)B * 128 * 128 * 256), TPB, 0, stream>>>(x2, upb, cat, (long long)B * 128 * 128, 128, 128);
    spectral_block(stream, scratch, cat, u, up3.w1, up3.w2, up3.wr, up3.br, B, 128, 256, 64, 16, nullptr, nullptr, 0);

    k_upsample2<<<ewblocks((long long)B * 256 * 256 * 64), TPB, 0, stream>>>(u, upb, B, 128, 128, 64);
    k_concat<<<ewblocks((long long)B * 256 * 256 * 128), TPB, 0, stream>>>(x1, upb, cat, (long long)B * 256 * 256, 64, 64);
    spectral_block(stream, scratch, cat, u, up4.w1, up4.w2, up4.wr, up4.br, B, 256, 128, 64, 32, nullptr, nullptr, 0);

    // ---- output head ----
    launch_gemm(stream, u, IN(38), IN(39), (float*)d_out,
                B * 256 * 256, 4, 64, 64, 4, 4, 0, 0, 0, 1, 0, 0);
#undef IN
}
